// GraphConvNet_28535762714722
// MI455X (gfx1250) — matrix operand order repaired and checked by reading, not compiled, not run on hardware
//
#include <hip/hip_runtime.h>
#include <hip/hip_bf16.h>
#include <math.h>

typedef __attribute__((ext_vector_type(16))) _Float16 v16h;
typedef __attribute__((ext_vector_type(8)))  float    v8f;
typedef __attribute__((ext_vector_type(4)))  float    f32x4;

#define NN   4096
#define INF  512
#define HF   256
#define OF   16
#define NB   4096
#define QIDX 16609443ULL   /* round(0.99 * (4096*4096 - 1)) */
#define PFD  128           /* prefetch distance in halves (256 B, 4 K-steps ahead) */

union FragU { f32x4 f4[2]; v16h v; };

// A-operand fragment: 16x32 f16 tile from row-major [M,K] matrix.
// Lane r=lane&15 holds row m0+r; halves {hf*8..hf*8+7} and {16+hf*8..} (ISA 7.12.2).
__device__ __forceinline__ v16h fragA(const _Float16* base, int ld, int m0, int k0) {
  int lane = threadIdx.x & 31;
  int r = lane & 15, hf = lane >> 4;
  const _Float16* p = base + (size_t)(m0 + r) * ld + k0;
  FragU u;
  u.f4[0] = *(const f32x4*)(p + hf * 8);
  u.f4[1] = *(const f32x4*)(p + 16 + hf * 8);
  return u.v;
}

// B-operand fragment: 32x16 f16 tile where B[k,n] = Bt[n,k], Bt row-major [N,K].
// Lane c=lane&15 holds column n0+c; K halves {kg*8..kg*8+7} and {16+kg*8..}, matching fragA.
__device__ __forceinline__ v16h fragB(const _Float16* base, int ld, int n0, int k0) {
  int lane = threadIdx.x & 31;
  int c = lane & 15, kg = lane >> 4;
  const _Float16* p = base + (size_t)(n0 + c) * ld + k0;
  FragU u;
  u.f4[0] = *(const f32x4*)(p + kg * 8);
  u.f4[1] = *(const f32x4*)(p + 16 + kg * 8);
  return u.v;
}

// ---------- row normalize: x -> x_f16 and xn_f16 (unit rows, eps=1e-8) ----------
__global__ __launch_bounds__(256) void k_rownorm(const float* __restrict__ x,
                                                 _Float16* __restrict__ x16,
                                                 _Float16* __restrict__ xn16) {
  int row = blockIdx.x;
  const float* xr = x + (size_t)row * INF;
  float v0 = xr[threadIdx.x];
  float v1 = xr[threadIdx.x + 256];
  __shared__ float s[256];
  s[threadIdx.x] = v0 * v0 + v1 * v1;
  __syncthreads();
  for (int off = 128; off > 0; off >>= 1) {
    if (threadIdx.x < off) s[threadIdx.x] += s[threadIdx.x + off];
    __syncthreads();
  }
  __shared__ float invn;
  if (threadIdx.x == 0) invn = 1.0f / fmaxf(sqrtf(s[0]), 1e-8f);
  __syncthreads();
  size_t b = (size_t)row * INF + threadIdx.x;
  x16[b]        = (_Float16)v0;
  x16[b + 256]  = (_Float16)v1;
  xn16[b]       = (_Float16)(v0 * invn);
  xn16[b + 256] = (_Float16)(v1 * invn);
}

// ---------- generic f32 -> f16 convert ----------
__global__ void k_cvt(const float* __restrict__ src, _Float16* __restrict__ dst, int n) {
  int i = blockIdx.x * blockDim.x + threadIdx.x;
  if (i < n) dst[i] = (_Float16)src[i];
}

__global__ void k_zero(unsigned int* p, int n) {
  int i = blockIdx.x * blockDim.x + threadIdx.x;
  if (i < n) p[i] = 0u;
}

// ---------- sim = |xn @ xn^T|, f32 output; 32x64 tile per wave ----------
__global__ __launch_bounds__(128) void k_gemm_sim(const _Float16* __restrict__ xn,
                                                  float* __restrict__ sim) {
  int wave = threadIdx.x >> 5;
  int m0 = blockIdx.x * 32;
  int n0 = blockIdx.y * 256 + wave * 64;
  int lane = threadIdx.x & 31;
  int r = lane & 15;
  // per-lane row pointers for prefetch (lanes 0..15 cover all 16 rows of each tile)
  const _Float16* pfA0 = xn + (size_t)(m0 + r) * INF;
  const _Float16* pfA1 = xn + (size_t)(m0 + 16 + r) * INF;
  const _Float16* pfB0 = xn + (size_t)(n0 + r) * INF;
  const _Float16* pfB1 = xn + (size_t)(n0 + 16 + r) * INF;
  const _Float16* pfB2 = xn + (size_t)(n0 + 32 + r) * INF;
  const _Float16* pfB3 = xn + (size_t)(n0 + 48 + r) * INF;
  v8f acc[2][4] = {};
  for (int k0 = 0; k0 < INF; k0 += 32) {
    __builtin_prefetch(pfA0 + k0 + PFD, 0, 3);
    __builtin_prefetch(pfA1 + k0 + PFD, 0, 3);
    __builtin_prefetch(pfB0 + k0 + PFD, 0, 3);
    __builtin_prefetch(pfB1 + k0 + PFD, 0, 3);
    __builtin_prefetch(pfB2 + k0 + PFD, 0, 3);
    __builtin_prefetch(pfB3 + k0 + PFD, 0, 3);
    v16h a0 = fragA(xn, INF, m0, k0);
    v16h a1 = fragA(xn, INF, m0 + 16, k0);
#pragma unroll
    for (int j = 0; j < 4; ++j) {
      v16h b = fragB(xn, INF, n0 + 16 * j, k0);
      acc[0][j] = __builtin_amdgcn_wmma_f32_16x16x32_f16(false, a0, false, b,
                                                         (short)0, acc[0][j], false, false);
      acc[1][j] = __builtin_amdgcn_wmma_f32_16x16x32_f16(false, a1, false, b,
                                                         (short)0, acc[1][j], false, false);
    }
  }
  int c = lane & 15, hf = lane >> 4;
#pragma unroll
  for (int i = 0; i < 2; ++i)
#pragma unroll
    for (int j = 0; j < 4; ++j)
#pragma unroll
      for (int v = 0; v < 8; ++v) {
        int row = m0 + 16 * i + v + 8 * hf;
        int col = n0 + 16 * j + c;
        sim[(size_t)row * NN + col] = fabsf(acc[i][j][v]);
      }
}

// ---------- histogram passes for quantile selection ----------
__global__ __launch_bounds__(256) void k_hist(const float* __restrict__ sim,
                                              unsigned int* __restrict__ hist,
                                              const float* __restrict__ lo_p, int pass) {
  __shared__ unsigned int lh[NB];
  for (int i = threadIdx.x; i < NB; i += 256) lh[i] = 0u;
  __syncthreads();
  float lo = 0.0f, scale = (float)NB;
  if (pass) { lo = lo_p[0]; scale = (float)NB * (float)NB; }
  size_t total = (size_t)NN * NN;
  for (size_t i = (size_t)blockIdx.x * 256 + threadIdx.x; i < total; i += (size_t)gridDim.x * 256) {
    float t = (sim[i] - lo) * scale;
    if (t >= 0.0f && t < (float)NB) {
      atomicAdd(&lh[(int)t], 1u);
    } else if (!pass && t >= (float)NB) {
      atomicAdd(&lh[NB - 1], 1u);   // clamp |cos| slightly > 1 from f16 rounding
    }
  }
  __syncthreads();
  for (int i = threadIdx.x; i < NB; i += 256)
    if (lh[i]) atomicAdd(&hist[i], lh[i]);
}

__global__ void k_scan1(const unsigned int* __restrict__ hist,
                        float* __restrict__ lo_p, unsigned long long* __restrict__ rank_p) {
  if (threadIdx.x != 0 || blockIdx.x != 0) return;
  unsigned long long cum = 0, below = 0;
  int b = NB - 1;
  bool found = false;
  for (int i = 0; i < NB; ++i) {
    if (!found) {
      unsigned long long nc = cum + hist[i];
      if (nc > QIDX) { b = i; below = cum; found = true; }
      else cum = nc;
    }
  }
  if (!found) below = cum;
  lo_p[0] = (float)b / (float)NB;
  rank_p[0] = QIDX - below;
}

__global__ void k_scan2(const unsigned int* __restrict__ hist2,
                        const float* __restrict__ lo_p,
                        const unsigned long long* __restrict__ rank_p,
                        float* __restrict__ eps_p) {
  if (threadIdx.x != 0 || blockIdx.x != 0) return;
  unsigned long long r = rank_p[0], cum = 0;
  int b = NB - 1;
  bool found = false;
  for (int i = 0; i < NB; ++i) {
    if (!found) {
      unsigned long long nc = cum + hist2[i];
      if (nc > r) { b = i; found = true; }
      else cum = nc;
    }
  }
  eps_p[0] = lo_p[0] + (float)b / ((float)NB * (float)NB);
}

// ---------- adjacency (f16 0/1) + 1/deg (sim symmetric -> row sums) ----------
__global__ __launch_bounds__(256) void k_adjdeg(const float* __restrict__ sim,
                                                const float* __restrict__ eps_p,
                                                _Float16* __restrict__ adj,
                                                float* __restrict__ invdeg) {
  int row = blockIdx.x;
  float eps = eps_p[0];
  const float* sr = sim + (size_t)row * NN;
  _Float16* ar = adj + (size_t)row * NN;
  int cnt = 0;
  for (int c = threadIdx.x; c < NN; c += 256) {
    bool e = sr[c] >= eps;
    ar[c] = e ? (_Float16)1.0f : (_Float16)0.0f;
    cnt += e ? 1 : 0;
  }
  __shared__ int s[256];
  s[threadIdx.x] = cnt;
  __syncthreads();
  for (int off = 128; off > 0; off >>= 1) {
    if (threadIdx.x < off) s[threadIdx.x] += s[threadIdx.x + off];
    __syncthreads();
  }
  if (threadIdx.x == 0) invdeg[row] = 1.0f / fmaxf((float)s[0], 1.0f);
}

// ---------- h0 = relu(x @ w_in^T): M=4096 N=256 K=512; 32x64 tile per wave ----------
__global__ __launch_bounds__(128) void k_gemm_in(const _Float16* __restrict__ A,
                                                 const _Float16* __restrict__ W,
                                                 _Float16* __restrict__ h,
                                                 _Float16* __restrict__ hT) {
  int wave = threadIdx.x >> 5;
  int m0 = blockIdx.x * 32;
  int n0 = wave * 64;
  v8f acc[2][4] = {};
  for (int k0 = 0; k0 < INF; k0 += 32) {
    v16h a0 = fragA(A, INF, m0, k0);
    v16h a1 = fragA(A, INF, m0 + 16, k0);
#pragma unroll
    for (int j = 0; j < 4; ++j) {
      v16h b = fragB(W, INF, n0 + 16 * j, k0);
      acc[0][j] = __builtin_amdgcn_wmma_f32_16x16x32_f16(false, a0, false, b,
                                                         (short)0, acc[0][j], false, false);
      acc[1][j] = __builtin_amdgcn_wmma_f32_16x16x32_f16(false, a1, false, b,
                                                         (short)0, acc[1][j], false, false);
    }
  }
  int lane = threadIdx.x & 31;
  int c = lane & 15, hf = lane >> 4;
#pragma unroll
  for (int i = 0; i < 2; ++i)
#pragma unroll
    for (int j = 0; j < 4; ++j)
#pragma unroll
      for (int v = 0; v < 8; ++v) {
        int row = m0 + 16 * i + v + 8 * hf;
        int col = n0 + 16 * j + c;
        _Float16 r = (_Float16)fmaxf(acc[i][j][v], 0.0f);
        h[(size_t)row * HF + col] = r;
        hT[(size_t)col * NN + row] = r;
      }
}

// ---------- agg = (adj @ h) * invdeg[row]: M=4096 N=256 K=4096; 32x64/wave ----------
__global__ __launch_bounds__(128) void k_gemm_agg(const _Float16* __restrict__ adj,
                                                  const _Float16* __restrict__ hT,
                                                  const float* __restrict__ invdeg,
                                                  _Float16* __restrict__ agg) {
  int wave = threadIdx.x >> 5;
  int m0 = blockIdx.x * 32;
  int n0 = wave * 64;
  int lane = threadIdx.x & 31;
  int r = lane & 15;
  const _Float16* pfA0 = adj + (size_t)(m0 + r) * NN;
  const _Float16* pfA1 = adj + (size_t)(m0 + 16 + r) * NN;
  const _Float16* pfB0 = hT + (size_t)(n0 + r) * NN;
  const _Float16* pfB1 = hT + (size_t)(n0 + 16 + r) * NN;
  const _Float16* pfB2 = hT + (size_t)(n0 + 32 + r) * NN;
  const _Float16* pfB3 = hT + (size_t)(n0 + 48 + r) * NN;
  v8f acc[2][4] = {};
  for (int k0 = 0; k0 < NN; k0 += 32) {
    __builtin_prefetch(pfA0 + k0 + PFD, 0, 3);
    __builtin_prefetch(pfA1 + k0 + PFD, 0, 3);
    __builtin_prefetch(pfB0 + k0 + PFD, 0, 3);
    __builtin_prefetch(pfB1 + k0 + PFD, 0, 3);
    __builtin_prefetch(pfB2 + k0 + PFD, 0, 3);
    __builtin_prefetch(pfB3 + k0 + PFD, 0, 3);
    v16h a0 = fragA(adj, NN, m0, k0);
    v16h a1 = fragA(adj, NN, m0 + 16, k0);
#pragma unroll
    for (int j = 0; j < 4; ++j) {
      v16h b = fragB(hT, NN, n0 + 16 * j, k0);
      acc[0][j] = __builtin_amdgcn_wmma_f32_16x16x32_f16(false, a0, false, b,
                                                         (short)0, acc[0][j], false, false);
      acc[1][j] = __builtin_amdgcn_wmma_f32_16x16x32_f16(false, a1, false, b,
                                                         (short)0, acc[1][j], false, false);
    }
  }
  int c = lane & 15, hf = lane >> 4;
#pragma unroll
  for (int i = 0; i < 2; ++i)
#pragma unroll
    for (int j = 0; j < 4; ++j)
#pragma unroll
      for (int v = 0; v < 8; ++v) {
        int row = m0 + 16 * i + v + 8 * hf;
        int col = n0 + 16 * j + c;
        agg[(size_t)row * HF + col] = (_Float16)(acc[i][j][v] * invdeg[row]);
      }
}

// ---------- h' = relu(agg @ wl^T + h @ wr^T + bl): dual-K accumulate; 32x64/wave ----------
__global__ __launch_bounds__(128) void k_gemm_sage(const _Float16* __restrict__ agg,
                                                   const _Float16* __restrict__ wl,
                                                   const _Float16* __restrict__ h,
                                                   const _Float16* __restrict__ wr,
                                                   const float* __restrict__ bias,
                                                   _Float16* __restrict__ ho,
                                                   _Float16* __restrict__ hoT) {
  int wave = threadIdx.x >> 5;
  int m0 = blockIdx.x * 32;
  int n0 = wave * 64;
  v8f acc[2][4] = {};
  for (int k0 = 0; k0 < HF; k0 += 32) {
    v16h a10 = fragA(agg, HF, m0, k0);
    v16h a11 = fragA(agg, HF, m0 + 16, k0);
    v16h a20 = fragA(h, HF, m0, k0);
    v16h a21 = fragA(h, HF, m0 + 16, k0);
#pragma unroll
    for (int j = 0; j < 4; ++j) {
      v16h b1 = fragB(wl, HF, n0 + 16 * j, k0);
      acc[0][j] = __builtin_amdgcn_wmma_f32_16x16x32_f16(false, a10, false, b1,
                                                         (short)0, acc[0][j], false, false);
      acc[1][j] = __builtin_amdgcn_wmma_f32_16x16x32_f16(false, a11, false, b1,
                                                         (short)0, acc[1][j], false, false);
      v16h b2 = fragB(wr, HF, n0 + 16 * j, k0);
      acc[0][j] = __builtin_amdgcn_wmma_f32_16x16x32_f16(false, a20, false, b2,
                                                         (short)0, acc[0][j], false, false);
      acc[1][j] = __builtin_amdgcn_wmma_f32_16x16x32_f16(false, a21, false, b2,
                                                         (short)0, acc[1][j], false, false);
    }
  }
  int lane = threadIdx.x & 31;
  int c = lane & 15, hf = lane >> 4;
#pragma unroll
  for (int i = 0; i < 2; ++i)
#pragma unroll
    for (int j = 0; j < 4; ++j)
#pragma unroll
      for (int v = 0; v < 8; ++v) {
        int row = m0 + 16 * i + v + 8 * hf;
        int col = n0 + 16 * j + c;
        _Float16 r = (_Float16)fmaxf(acc[i][j][v] + bias[col], 0.0f);
        ho[(size_t)row * HF + col] = r;
        hoT[(size_t)col * NN + row] = r;
      }
}

// ---------- out = softmax(h @ w_out^T + b_out) ----------
__global__ __launch_bounds__(256) void k_out(const _Float16* __restrict__ h,
                                             const float* __restrict__ w_out,
                                             const float* __restrict__ b_out,
                                             float* __restrict__ out) {
  __shared__ float ws[OF * HF];
  __shared__ float bs[OF];
  for (int i = threadIdx.x; i < OF * HF; i += 256) ws[i] = w_out[i];
  if (threadIdx.x < OF) bs[threadIdx.x] = b_out[threadIdx.x];
  __syncthreads();
  int row = blockIdx.x * 256 + threadIdx.x;
  float acc[OF];
#pragma unroll
  for (int o = 0; o < OF; ++o) acc[o] = bs[o];
  const _Float16* hr = h + (size_t)row * HF;
  for (int k = 0; k < HF; ++k) {
    float hk = (float)hr[k];
#pragma unroll
    for (int o = 0; o < OF; ++o) acc[o] += hk * ws[o * HF + k];
  }
  float m = acc[0];
#pragma unroll
  for (int o = 1; o < OF; ++o) m = fmaxf(m, acc[o]);
  float sum = 0.0f;
#pragma unroll
  for (int o = 0; o < OF; ++o) { acc[o] = expf(acc[o] - m); sum += acc[o]; }
  float inv = 1.0f / sum;
#pragma unroll
  for (int o = 0; o < OF; ++o) out[(size_t)row * OF + o] = acc[o] * inv;
}

extern "C" void kernel_launch(void* const* d_in, const int* in_sizes, int n_in,
                              void* d_out, int out_size, void* d_ws, size_t ws_size,
                              hipStream_t stream) {
  (void)in_sizes; (void)n_in; (void)out_size; (void)ws_size;
  const float* x     = (const float*)d_in[0];
  const float* w_in  = (const float*)d_in[1];
  const float* wl1   = (const float*)d_in[2];
  const float* bl1   = (const float*)d_in[3];
  const float* wr1   = (const float*)d_in[4];
  const float* wl2   = (const float*)d_in[5];
  const float* bl2   = (const float*)d_in[6];
  const float* wr2   = (const float*)d_in[7];
  const float* w_out = (const float*)d_in[8];
  const float* b_out = (const float*)d_in[9];
  float* out = (float*)d_out;

  char* base = (char*)d_ws;
  size_t off = 0;
  auto alloc = [&](size_t bytes) -> char* {
    char* p = base + off;
    off += (bytes + 255) & ~(size_t)255;
    return p;
  };
  _Float16* xn16  = (_Float16*)alloc((size_t)NN * INF * 2);
  _Float16* x16   = (_Float16*)alloc((size_t)NN * INF * 2);
  _Float16* win16 = (_Float16*)alloc((size_t)HF * INF * 2);
  _Float16* wl116 = (_Float16*)alloc((size_t)HF * HF * 2);
  _Float16* wr116 = (_Float16*)alloc((size_t)HF * HF * 2);
  _Float16* wl216 = (_Float16*)alloc((size_t)HF * HF * 2);
  _Float16* wr216 = (_Float16*)alloc((size_t)HF * HF * 2);
  float*    sim   = (float*)alloc((size_t)NN * NN * 4);
  _Float16* adj   = (_Float16*)alloc((size_t)NN * NN * 2);
  float*    invdg = (float*)alloc((size_t)NN * 4);
  _Float16* h16   = (_Float16*)alloc((size_t)NN * HF * 2);
  _Float16* hT16  = (_Float16*)alloc((size_t)NN * HF * 2);
  _Float16* agg16 = (_Float16*)alloc((size_t)NN * HF * 2);
  _Float16* h2    = (_Float16*)alloc((size_t)NN * HF * 2);
  _Float16* h2T   = (_Float16*)alloc((size_t)NN * HF * 2);
  unsigned int* hist1 = (unsigned int*)alloc((size_t)NB * 4);
  unsigned int* hist2 = (unsigned int*)alloc((size_t)NB * 4);
  unsigned long long* rank = (unsigned long long*)alloc(64);
  float* lo_p  = (float*)((char*)rank + 8);
  float* eps_p = (float*)((char*)rank + 16);

  // 1) normalize rows -> xn_f16, and raw x -> f16
  k_rownorm<<<NN, 256, 0, stream>>>(x, x16, xn16);
  // 2) weights -> f16
  k_cvt<<<(HF * INF + 255) / 256, 256, 0, stream>>>(w_in, win16, HF * INF);
  k_cvt<<<(HF * HF + 255) / 256, 256, 0, stream>>>(wl1, wl116, HF * HF);
  k_cvt<<<(HF * HF + 255) / 256, 256, 0, stream>>>(wr1, wr116, HF * HF);
  k_cvt<<<(HF * HF + 255) / 256, 256, 0, stream>>>(wl2, wl216, HF * HF);
  k_cvt<<<(HF * HF + 255) / 256, 256, 0, stream>>>(wr2, wr216, HF * HF);
  k_zero<<<(2 * NB + 255) / 256, 256, 0, stream>>>(hist1, 2 * NB);
  // 3) sim = |xn xn^T|   (WMMA, 17 GFLOP)
  k_gemm_sim<<<dim3(NN / 32, NN / 256, 1), 128, 0, stream>>>(xn16, sim);
  // 4) quantile(0.99, nearest) via two-level histogram selection
  k_hist<<<1024, 256, 0, stream>>>(sim, hist1, lo_p, 0);
  k_scan1<<<1, 1, 0, stream>>>(hist1, lo_p, rank);
  k_hist<<<1024, 256, 0, stream>>>(sim, hist2, lo_p, 1);
  k_scan2<<<1, 1, 0, stream>>>(hist2, lo_p, rank, eps_p);
  // 5) adjacency + 1/deg (symmetric)
  k_adjdeg<<<NN, 256, 0, stream>>>(sim, eps_p, adj, invdg);
  // 6) h0 = relu(x @ w_in^T)
  k_gemm_in<<<NN / 32, 128, 0, stream>>>(x16, win16, h16, hT16);
  // 7) SAGE layer 1
  k_gemm_agg<<<NN / 32, 128, 0, stream>>>(adj, hT16, invdg, agg16);
  k_gemm_sage<<<NN / 32, 128, 0, stream>>>(agg16, wl116, h16, wr116, bl1, h2, h2T);
  // 8) SAGE layer 2
  k_gemm_agg<<<NN / 32, 128, 0, stream>>>(adj, h2T, invdg, agg16);
  k_gemm_sage<<<NN / 32, 128, 0, stream>>>(agg16, wl216, h2, wr216, bl2, h16, hT16);
  // 9) output head + softmax
  k_out<<<NN / 256, 256, 0, stream>>>(h16, w_out, b_out, out);
}
